// GroupedQueryAttention_66571993088384
// MI455X (gfx1250) — matrix operand order repaired and checked by reading, not compiled, not run on hardware
//
#include <hip/hip_runtime.h>
#include <hip/hip_bf16.h>

typedef __attribute__((ext_vector_type(16))) __bf16 v16bf;
typedef __attribute__((ext_vector_type(8)))  float  v8f;
typedef __attribute__((ext_vector_type(4)))  unsigned int u32x4;
typedef __attribute__((ext_vector_type(8)))  unsigned int u32x8;

#define BATCH     2
#define SEQ_T     2048
#define D_MODEL   2048
#define NUM_HEADS 32
#define NUM_KV    8
#define DH        64

union FragBF { v16bf v; uint4 q[2]; __bf16 e[16]; };
union Pack32 { uint4 q[4]; __bf16 e[32]; };

__device__ __forceinline__ v8f zero_v8f() {
  v8f z = {0.f, 0.f, 0.f, 0.f, 0.f, 0.f, 0.f, 0.f};
  return z;
}

// LDS byte offset of a __shared__ pointer (low 32 bits of the LDS-aperture addr).
__device__ __forceinline__ unsigned lds_off(const void* p) {
  return (unsigned)(uintptr_t)p;
}

// Async global -> LDS, 16 bytes per lane (ASYNCcnt). CDNA5 VGLOBAL op 98.
__device__ __forceinline__ void async_load_b128(unsigned lds_addr, const void* gptr) {
  asm volatile("global_load_async_to_lds_b128 %0, %1, off"
               :: "v"(lds_addr), "v"((unsigned long long)(uintptr_t)gptr)
               : "memory");
}

__device__ __forceinline__ void wait_asynccnt0() {
  asm volatile("s_wait_asynccnt 0x0" ::: "memory");
}

// A-fragment (16x32 bf16, MxK). Lane l: row m = l&15; half = l>>4.
// elems 0..7  = K[half*8 .. half*8+7], elems 8..15 = K[16+half*8 .. 16+half*8+7].
__device__ __forceinline__ v16bf load_a_frag(const __bf16* __restrict__ base, int ldm,
                                             int row0, int k0, int lane) {
  const int m = lane & 15, half = lane >> 4;
  FragBF f;
  const __bf16* p = base + (size_t)(row0 + m) * ldm + k0 + half * 8;
  f.q[0] = *reinterpret_cast<const uint4*>(p);
  f.q[1] = *reinterpret_cast<const uint4*>(p + 16);
  return f.v;
}

// B-fragment (32x16 bf16, KxN). Lane l: col n = l&15; half = l>>4.
// elems 0..7  = K[half*8 .. half*8+7], elems 8..15 = K[16+half*8 .. 16+half*8+7].
__device__ __forceinline__ v16bf load_b_frag(const __bf16* __restrict__ base, int ldm,
                                             int k0, int n0, int lane) {
  const int n = lane & 15, half = lane >> 4;
  FragBF f;
  const __bf16* p = base + (size_t)(k0 + half * 8) * ldm + n0 + n;
#pragma unroll
  for (int s = 0; s < 8; ++s) {
    f.e[s] = p[(size_t)s * ldm];
    f.e[8 + s] = p[(size_t)(16 + s) * ldm];
  }
  return f.v;
}

// ---------------- fp32 -> bf16 conversion ----------------
__global__ void f32_to_bf16_kernel(const float* __restrict__ in, __bf16* __restrict__ out, int n) {
  int i = blockIdx.x * blockDim.x + threadIdx.x;
  int stride = gridDim.x * blockDim.x;
  for (; i < n; i += stride) out[i] = (__bf16)in[i];
}

// ---------------- bf16 GEMM: C[M,N] = A[M,K] @ B[K,N] ----------------
// Tile 128x64x32; 256 threads = 8 waves in a 4(M) x 2(N) grid; 32x32 per wave.
// A tile staged with async global->LDS; B tile staged with the Tensor Data Mover
// (wave 0 issues one descriptor per K-step; TDM inserts the LDS row padding).
template <bool OUT_F32>
__global__ __launch_bounds__(256) void gemm_bf16_kernel(
    const __bf16* __restrict__ A, const __bf16* __restrict__ B,
    void* __restrict__ Cv, int M, int N, int K) {
  __shared__ alignas(16) __bf16 Ash[128][40];  // +8 pad, rows 16B aligned
  __shared__ alignas(16) __bf16 Bsh[32][72];   // +8 pad (TDM pad: 32 dw + 4 dw)
  const int tid  = threadIdx.x;
  const int lane = tid & 31, wid = tid >> 5;
  const int wm = (wid & 3) * 32, wn = (wid >> 2) * 32;
  const int blockM = blockIdx.y * 128, blockN = blockIdx.x * 64;

  v8f acc[2][2] = { {zero_v8f(), zero_v8f()}, {zero_v8f(), zero_v8f()} };

  const int arow = tid >> 2, acol = (tid & 3) * 8;  // 64 rows/pass, 2 passes

  for (int k0 = 0; k0 < K; k0 += 32) {
    // A tile: 128x32 bf16 via async global->LDS (16B per lane, 2 passes)
#pragma unroll
    for (int pass = 0; pass < 2; ++pass) {
      int r = arow + pass * 64;
      async_load_b128(lds_off(&Ash[r][acol]),
                      &A[(size_t)(blockM + r) * K + k0 + acol]);
    }
    // B tile: 32x64 bf16 via TDM (wave 0 only; descriptor is wave-uniform)
    if (wid == 0) {
      unsigned long long ga =
          (unsigned long long)(uintptr_t)(&B[(size_t)k0 * N + blockN]);
      u32x4 g0;
      g0.x = 1u;                                   // count=1 (valid descriptor)
      g0.y = lds_off(&Bsh[0][0]);                  // lds_addr
      g0.z = (unsigned)(ga & 0xffffffffu);         // global_addr[31:0]
      g0.w = (unsigned)((ga >> 32) & 0x01ffffffu)  // global_addr[56:32]
             | (2u << 30);                         // type = 2 ("image")
      u32x8 g1;
      g1[0] = (1u << 16)                           // data_size = 2 bytes
            | (1u << 20)                           // pad_enable
            | (4u << 22)                           // pad_interval: 32 DWORDs (128B row)
            | (3u << 25);                          // pad_amount: 4 DWORDs (+16B)
      g1[1] = ((unsigned)N & 0xffffu) << 16;       // tensor_dim0[15:0]
      g1[2] = ((unsigned)N >> 16)                  // tensor_dim0[31:16]
            | (((unsigned)K & 0xffffu) << 16);     // tensor_dim1[15:0]
      g1[3] = ((unsigned)K >> 16) | (64u << 16);   // tensor_dim1[31:16], tile_dim0=64
      g1[4] = 32u;                                 // tile_dim1=32, tile_dim2=0
      g1[5] = (unsigned)N;                         // tensor_dim0_stride[31:0]
      g1[6] = 0u;                                  // stride hi / dim1_stride lo
      g1[7] = 0u;
      asm volatile("tensor_load_to_lds %0, %1" :: "s"(g0), "s"(g1) : "memory");
      __builtin_amdgcn_s_wait_tensorcnt(0);
    }
    wait_asynccnt0();
    __syncthreads();

    v16bf af[2], bfg[2];
#pragma unroll
    for (int i = 0; i < 2; ++i) af[i]  = load_a_frag(&Ash[0][0], 40, wm + i * 16, 0, lane);
#pragma unroll
    for (int j = 0; j < 2; ++j) bfg[j] = load_b_frag(&Bsh[0][0], 72, 0, wn + j * 16, lane);
#pragma unroll
    for (int i = 0; i < 2; ++i)
#pragma unroll
      for (int j = 0; j < 2; ++j)
        acc[i][j] = __builtin_amdgcn_wmma_f32_16x16x32_bf16(
            false, af[i], false, bfg[j], (short)0, acc[i][j], false, false);
    __syncthreads();
  }

  const int half = lane >> 4, nl = lane & 15;
#pragma unroll
  for (int i = 0; i < 2; ++i)
#pragma unroll
    for (int j = 0; j < 2; ++j)
#pragma unroll
      for (int r = 0; r < 8; ++r) {
        int row = blockM + wm + i * 16 + r + half * 8;
        int col = blockN + wn + j * 16 + nl;
        float val = acc[i][j][r];
        if constexpr (OUT_F32)
          reinterpret_cast<float*>(Cv)[(size_t)row * N + col] = val;
        else
          reinterpret_cast<__bf16*>(Cv)[(size_t)row * N + col] = (__bf16)val;
      }
}

// ---------------- Flash-attention (causal, GQA) ----------------
// Grid: (T/64, H, B). Block: 128 threads = 4 waves; wave w owns q-rows [w*16, w*16+16).
__global__ __launch_bounds__(128) void gqa_attn_kernel(
    const __bf16* __restrict__ Q, const __bf16* __restrict__ K,
    const __bf16* __restrict__ V, __bf16* __restrict__ O) {
  __shared__ alignas(16) __bf16 Qsh[64][72];
  __shared__ alignas(16) __bf16 KshT[64][72];   // transposed: [d][key]
  __shared__ alignas(16) __bf16 Vsh[64][72];    // [key][d]
  __shared__ alignas(16) float  Ssh[4][16][68];
  __shared__ alignas(16) __bf16 Psh[4][16][72];
  __shared__ float alpha_sh[4][16];
  __shared__ float rsum_sh[4][16];

  const int tid  = threadIdx.x;
  const int lane = tid & 31, w = tid >> 5;
  const int half = lane >> 4, nl = lane & 15;
  const int qt = blockIdx.x, h = blockIdx.y, b = blockIdx.z;
  const int hk = h >> 2;  // REPS = 4
  const int q0 = qt * 64;

  // Stage Q tile (async): 64 rows x 64 dh, row stride H*DH in global
  {
    const int r = tid >> 1, dbase = (tid & 1) * 32;
    const __bf16* src = Q + ((size_t)(b * SEQ_T + q0 + r) * NUM_HEADS + h) * DH + dbase;
#pragma unroll
    for (int i = 0; i < 4; ++i)
      async_load_b128(lds_off(&Qsh[r][dbase + i * 8]), src + i * 8);
  }

  float row_max = -__builtin_inff();  // lanes 0..15 carry state for their row
  float row_sum = 0.f;
  v8f Oacc[4] = { zero_v8f(), zero_v8f(), zero_v8f(), zero_v8f() };

  for (int kt = 0; kt <= qt; ++kt) {
    __syncthreads();  // previous tile fully consumed before restaging
    {
      const int key = tid >> 1, dbase = (tid & 1) * 32;
      const size_t kv_off =
          ((size_t)(b * SEQ_T + kt * 64 + key) * NUM_KV + hk) * DH + dbase;
      // V tile straight into LDS via async copies
#pragma unroll
      for (int i = 0; i < 4; ++i)
        async_load_b128(lds_off(&Vsh[key][dbase + i * 8]), V + kv_off + i * 8);
      // K tile needs a transpose: load to VGPRs, scatter 16-bit to KshT
      Pack32 kp;
#pragma unroll
      for (int i = 0; i < 4; ++i)
        kp.q[i] = *reinterpret_cast<const uint4*>(K + kv_off + i * 8);
#pragma unroll
      for (int e = 0; e < 32; ++e) KshT[dbase + e][key] = kp.e[e];
    }
    wait_asynccnt0();  // covers V this iter (+ Q on first iter)
    __syncthreads();

    // S = Q_w (16x64) @ K^T (64x64): accumulate over dh chunks of 32
    v8f S[4] = { zero_v8f(), zero_v8f(), zero_v8f(), zero_v8f() };
#pragma unroll
    for (int kc = 0; kc < 2; ++kc) {
      v16bf a = load_a_frag(&Qsh[0][0], 72, w * 16, kc * 32, lane);
#pragma unroll
      for (int nt = 0; nt < 4; ++nt) {
        v16bf bf = load_b_frag(&KshT[0][0], 72, kc * 32, nt * 16, lane);
        S[nt] = __builtin_amdgcn_wmma_f32_16x16x32_bf16(
            false, a, false, bf, (short)0, S[nt], false, false);
      }
    }
    const float scale = 0.125f;  // 1/sqrt(64)
#pragma unroll
    for (int nt = 0; nt < 4; ++nt)
#pragma unroll
      for (int r = 0; r < 8; ++r)
        Ssh[w][r + half * 8][nt * 16 + nl] = S[nt][r] * scale;
    __syncthreads();

    // Online softmax: lane l < 16 owns row m = l of this wave's tile
    if (lane < 16) {
      const int m = lane;
      const int qg = q0 + w * 16 + m;
      int nvalid = qg - kt * 64 + 1;
      if (nvalid > 64) nvalid = 64;
      float mt = -__builtin_inff();
      for (int n = 0; n < nvalid; ++n) mt = fmaxf(mt, Ssh[w][m][n]);
      float nm = fmaxf(row_max, mt);
      float al = __expf(row_max - nm);
      float s = 0.f;
      for (int n = 0; n < 64; ++n) {
        float p = (n < nvalid) ? __expf(Ssh[w][m][n] - nm) : 0.f;
        Psh[w][m][n] = (__bf16)p;
        s += p;
      }
      row_sum = row_sum * al + s;
      row_max = nm;
      alpha_sh[w][m] = al;
    }
    __syncthreads();

    // Rescale O by alpha (per row), then O += P @ V
    float alv[8];
#pragma unroll
    for (int r = 0; r < 8; ++r) alv[r] = alpha_sh[w][r + half * 8];
#pragma unroll
    for (int nt = 0; nt < 4; ++nt)
#pragma unroll
      for (int r = 0; r < 8; ++r) Oacc[nt][r] *= alv[r];

#pragma unroll
    for (int kc = 0; kc < 2; ++kc) {
      v16bf a = load_a_frag(&Psh[w][0][0], 72, 0, kc * 32, lane);
#pragma unroll
      for (int nt = 0; nt < 4; ++nt) {
        v16bf bf = load_b_frag(&Vsh[0][0], 72, kc * 32, nt * 16, lane);
        Oacc[nt] = __builtin_amdgcn_wmma_f32_16x16x32_bf16(
            false, a, false, bf, (short)0, Oacc[nt], false, false);
      }
    }
  }

  if (lane < 16) rsum_sh[w][lane] = 1.f / row_sum;
  __syncthreads();
  float inv[8];
#pragma unroll
  for (int r = 0; r < 8; ++r) inv[r] = rsum_sh[w][r + half * 8];
#pragma unroll
  for (int nt = 0; nt < 4; ++nt)
#pragma unroll
    for (int r = 0; r < 8; ++r) {
      int qg = q0 + w * 16 + r + half * 8;
      int d  = nt * 16 + nl;
      O[((size_t)(b * SEQ_T + qg) * NUM_HEADS + h) * DH + d] =
          (__bf16)(Oacc[nt][r] * inv[r]);
    }
}

// ---------------- launcher ----------------
extern "C" void kernel_launch(void* const* d_in, const int* in_sizes, int n_in,
                              void* d_out, int out_size, void* d_ws, size_t ws_size,
                              hipStream_t stream) {
  (void)in_sizes; (void)n_in; (void)out_size; (void)ws_size;
  const float* x  = (const float*)d_in[0];
  const float* wq = (const float*)d_in[1];
  const float* wk = (const float*)d_in[2];
  const float* wv = (const float*)d_in[3];
  const float* wo = (const float*)d_in[4];
  float* out = (float*)d_out;

  const int M   = BATCH * SEQ_T;                 // 4096
  const int n_x  = M * D_MODEL;                  // 8,388,608
  const int n_wq = D_MODEL * D_MODEL;            // 4,194,304
  const int n_wk = D_MODEL * (NUM_KV * DH);      // 1,048,576
  const int n_q  = M * (NUM_HEADS * DH);         // 8,388,608
  const int n_kv = M * (NUM_KV * DH);            // 2,097,152

  char* ws = (char*)d_ws;
  size_t off = 0;
  auto carve = [&](size_t bytes) {
    void* p = ws + off;
    off += (bytes + 255) & ~(size_t)255;
    return p;
  };
  __bf16* xb  = (__bf16*)carve((size_t)n_x  * 2);
  __bf16* wqb = (__bf16*)carve((size_t)n_wq * 2);
  __bf16* wkb = (__bf16*)carve((size_t)n_wk * 2);
  __bf16* wvb = (__bf16*)carve((size_t)n_wk * 2);
  __bf16* wob = (__bf16*)carve((size_t)n_wq * 2);
  __bf16* Qb  = (__bf16*)carve((size_t)n_q  * 2);
  __bf16* Kb  = (__bf16*)carve((size_t)n_kv * 2);
  __bf16* Vb  = (__bf16*)carve((size_t)n_kv * 2);
  __bf16* Ab  = (__bf16*)carve((size_t)n_q  * 2);

  // 1) fp32 -> bf16
  f32_to_bf16_kernel<<<4096, 256, 0, stream>>>(x,  xb,  n_x);
  f32_to_bf16_kernel<<<4096, 256, 0, stream>>>(wq, wqb, n_wq);
  f32_to_bf16_kernel<<<2048, 256, 0, stream>>>(wk, wkb, n_wk);
  f32_to_bf16_kernel<<<2048, 256, 0, stream>>>(wv, wvb, n_wk);
  f32_to_bf16_kernel<<<4096, 256, 0, stream>>>(wo, wob, n_wq);

  // 2) Q/K/V projections (bf16 WMMA GEMMs)
  gemm_bf16_kernel<false><<<dim3(D_MODEL / 64, M / 128), 256, 0, stream>>>(
      xb, wqb, Qb, M, D_MODEL, D_MODEL);
  gemm_bf16_kernel<false><<<dim3((NUM_KV * DH) / 64, M / 128), 256, 0, stream>>>(
      xb, wkb, Kb, M, NUM_KV * DH, D_MODEL);
  gemm_bf16_kernel<false><<<dim3((NUM_KV * DH) / 64, M / 128), 256, 0, stream>>>(
      xb, wvb, Vb, M, NUM_KV * DH, D_MODEL);

  // 3) causal GQA flash attention
  gqa_attn_kernel<<<dim3(SEQ_T / 64, NUM_HEADS, BATCH), 128, 0, stream>>>(
      Qb, Kb, Vb, Ab);

  // 4) output projection -> fp32 result
  gemm_bf16_kernel<true><<<dim3(D_MODEL / 64, M / 128), 256, 0, stream>>>(
      Ab, wob, out, M, D_MODEL, D_MODEL);
}
